// GINLayer_2869038153787
// MI455X (gfx1250) — matrix operand order repaired and checked by reading, not compiled, not run on hardware
//
#include <hip/hip_runtime.h>
#include <hip/hip_bf16.h>

// ---------------------------------------------------------------------------
// Types for CDNA5 WMMA (wave32): v_wmma_f32_16x16x32_bf16
// ---------------------------------------------------------------------------
typedef __attribute__((ext_vector_type(16))) __bf16    v16bf;
typedef __attribute__((ext_vector_type(8)))  float     v8f;
typedef __attribute__((ext_vector_type(8)))  unsigned  v8u;

#define MLP_WAVES 4   // waves (16-node tiles) per block

// ---------------------------------------------------------------------------
// Split two floats into packed (2 x bf16) hi and lo dwords with x ~= hi + lo.
// TRUNCATION split: hi = x masked to bf16 (exact, 1 AND), lo = bf16(x - hi).
// The lo term absorbs the truncation residual, so combined accuracy is
// ~2^-16 relative (fp32-like), with pure integer packing (no RNE sequences,
// no 16-bit sub-register moves).
// ---------------------------------------------------------------------------
__device__ __forceinline__ void split_pack(float x0, float x1,
                                           unsigned& hi, unsigned& lo) {
    unsigned u0 = __builtin_bit_cast(unsigned, x0);
    unsigned u1 = __builtin_bit_cast(unsigned, x1);
    unsigned h0 = u0 & 0xffff0000u;
    unsigned h1 = u1 & 0xffff0000u;
    hi = h1 | (u0 >> 16);
    float r0 = x0 - __builtin_bit_cast(float, h0);
    float r1 = x1 - __builtin_bit_cast(float, h1);
    unsigned v0 = __builtin_bit_cast(unsigned, r0);
    unsigned v1 = __builtin_bit_cast(unsigned, r1);
    lo = (v1 & 0xffff0000u) | (v0 >> 16);
}

__device__ __forceinline__ v8f wmma_bf16(v16bf a, v16bf b, v8f c) {
    return __builtin_amdgcn_wmma_f32_16x16x32_bf16(
        /*neg_a=*/false, a, /*neg_b=*/false, b,
        /*c_mod=*/(short)0, c, /*reuse_a=*/false, /*reuse_b=*/false);
}
// Split-precision accumulate: C += A*B with A,B split into (hi,lo) bf16.
__device__ __forceinline__ v8f wmma_split(v16bf ah, v16bf al,
                                          v16bf bh, v16bf bl, v8f c) {
    c = wmma_bf16(al, bh, c);
    c = wmma_bf16(ah, bl, c);
    c = wmma_bf16(ah, bh, c);
    return c;
}

// ---------------------------------------------------------------------------
// Weight-fragment pre-pack.  One block (32 lanes) = one (kc, nt) fragment of
// a [K x 32] row-major weight matrix, emitted directly in the ISA B-matrix
// layout (32x16 bf16, ISA 7.12.2/7.12.4): lane L holds column N=L&15,
// element i -> K = 8*(L>>4) + i + ((i&8)?8:0), 2 bf16 per dword.
// Output layout per fragment: [hi:32 lanes x 8 dwords][lo:32 lanes x 8 dwords]
// = 512 dwords (2 KB).  fragIdx = kc*2 + nt.
// ---------------------------------------------------------------------------
__global__ void pack_wfrag_kernel(const float* __restrict__ W,   // [K x 32]
                                  unsigned* __restrict__ out) {
    const int kc = blockIdx.x >> 1, nt = blockIdx.x & 1;
    const int lane = threadIdx.x;
    const int ncol = lane & 15, khalf = lane >> 4;
    unsigned* hi = out + ((size_t)blockIdx.x * 2 + 0) * 256 + lane * 8;
    unsigned* lo = out + ((size_t)blockIdx.x * 2 + 1) * 256 + lane * 8;
#pragma unroll
    for (int p = 0; p < 8; ++p) {
        int k = 32 * kc + 8 * khalf + 2 * p + ((p & 4) ? 8 : 0);
        float x0 = W[(size_t)k * 32 + 16 * nt + ncol];
        float x1 = W[(size_t)(k + 1) * 32 + 16 * nt + ncol];
        unsigned h, l; split_pack(x0, x1, h, l);
        hi[p] = h; lo[p] = l;
    }
}

// Load a pre-packed B fragment (fragIdx within this weight's fragment array).
__device__ __forceinline__ void load_bfrag_packed(const unsigned* __restrict__ frag,
                                                  int fragIdx, int lane,
                                                  v16bf& bh, v16bf& bl) {
    const v8u* base = reinterpret_cast<const v8u*>(frag) + (size_t)fragIdx * 64 + lane;
    bh = __builtin_bit_cast(v16bf, base[0]);    // hi block: 32 v8u
    bl = __builtin_bit_cast(v16bf, base[32]);   // lo block: 32 v8u
}

// ---------------------------------------------------------------------------
// Utility kernels
// ---------------------------------------------------------------------------
__global__ void copy4_kernel(const float4* __restrict__ src,
                             float4* __restrict__ dst, long long n4) {
    long long i = (long long)blockIdx.x * blockDim.x + threadIdx.x;
    if (i < n4) dst[i] = src[i];
}

__global__ void zero_kernel(float* __restrict__ p, int n) {
    int i = blockIdx.x * blockDim.x + threadIdx.x;
    if (i < n) p[i] = 0.0f;
}

// ---------------------------------------------------------------------------
// Edge scatter: agg[dst] += x[src].  One thread = one (edge, 16-feature chunk).
// f32 atomics resolve in L2 (accumulator is L2-resident: <= 38 MB << 192 MB).
// ---------------------------------------------------------------------------
template<int D>
__global__ void edge_scatter_kernel(const float* __restrict__ X,
                                    const int* __restrict__ srcIdx,
                                    const int* __restrict__ dstIdx,
                                    float* __restrict__ agg, int nEdges) {
    const int CH = D / 16;
    long long tid = (long long)blockIdx.x * blockDim.x + threadIdx.x;
    long long total = (long long)nEdges * CH;
    if (tid >= total) return;
    int e  = (int)(tid / CH);
    int ch = (int)(tid % CH);
    int s = srcIdx[e], d = dstIdx[e];
    const float4* xr = reinterpret_cast<const float4*>(X + (size_t)s * D + ch * 16);
    float* ar = agg + (size_t)d * D + ch * 16;
#pragma unroll
    for (int q = 0; q < 4; ++q) {
        float4 v = xr[q];
        atomicAdd(ar + 4 * q + 0, v.x);
        atomicAdd(ar + 4 * q + 1, v.y);
        atomicAdd(ar + 4 * q + 2, v.z);
        atomicAdd(ar + 4 * q + 3, v.w);
    }
}

// ---------------------------------------------------------------------------
// Fused GIN MLP:  Y = relu( BN( relu( relu(X@Wa + ba) @ Wb + bb ) ) )
// One wave = one 16-node tile; both 16x16 output tiles (32 cols) per wave.
// WMMA 16x16x32 bf16 with hi/lo split for ~fp32 accuracy.
// A layout (16-bit A 16x32, ISA 7.12.2): lane L holds row M = L&15;
//   element i -> K = 8*(L>>4) + i + ((i&8)?8:0)  (pairs share a dword).
// ---------------------------------------------------------------------------
template<int KIN>
__global__ __launch_bounds__(MLP_WAVES * 32)
void gin_mlp_kernel(const float* __restrict__ X,        // [N, KIN]  (x + agg)
                    const unsigned* __restrict__ fragA, // packed Wa fragments
                    const float* __restrict__ ba,       // [32]
                    const unsigned* __restrict__ fragB, // packed Wb fragments
                    const float* __restrict__ bbias,    // [32]
                    const float* __restrict__ gg,       // BN gamma
                    const float* __restrict__ beta,     // BN beta
                    const float* __restrict__ mu,       // BN mean
                    const float* __restrict__ var,      // BN var
                    float* __restrict__ Y, int n)       // [N, 32]
{
    __shared__ float xs[MLP_WAVES][16][KIN];
    __shared__ float ys[MLP_WAVES][16][32];

    const int lane = threadIdx.x & 31;
    const int wave = threadIdx.x >> 5;
    const int tile = blockIdx.x * MLP_WAVES + wave;
    const int row0 = tile * 16;

    // ---- stage X tile to LDS (coalesced float4 loads; rows clamped so every
    // wave still runs the barriers on out-of-range tiles)
    {
        const int r  = lane >> 1;
        const int cb = (lane & 1) * (KIN / 2);
        int node = row0 + r;
        if (node >= n) node = n - 1;
        const float4* s4 = reinterpret_cast<const float4*>(X + (size_t)node * KIN + cb);
        float4* d4 = reinterpret_cast<float4*>(&xs[wave][r][cb]);
#pragma unroll
        for (int q = 0; q < KIN / 8; ++q) d4[q] = s4[q];
    }
    __syncthreads();

    const int mrow = lane & 15;
    const int hl   = lane >> 4;

    // ---- build A fragments (hi/lo split, dword-packed pairs)
    v16bf a1h[KIN / 32], a1l[KIN / 32];
#pragma unroll
    for (int kc = 0; kc < KIN / 32; ++kc) {
        v8u hu, lu;
#pragma unroll
        for (int p = 0; p < 8; ++p) {
            int i = 2 * p;
            int k = 32 * kc + 8 * hl + i + ((i & 8) ? 8 : 0);
            unsigned h, l;
            split_pack(xs[wave][mrow][k], xs[wave][mrow][k + 1], h, l);
            hu[p] = h; lu[p] = l;
        }
        a1h[kc] = __builtin_bit_cast(v16bf, hu);
        a1l[kc] = __builtin_bit_cast(v16bf, lu);
    }

    // ---- layer 1: (16 x KIN) @ (KIN x 32) -> two 16x16 tiles
    v8f c0 = {}, c1 = {};
#pragma unroll
    for (int kc = 0; kc < KIN / 32; ++kc) {
        v16bf bh, bl;
        load_bfrag_packed(fragA, kc * 2 + 0, lane, bh, bl);
        c0 = wmma_split(a1h[kc], a1l[kc], bh, bl, c0);
        load_bfrag_packed(fragA, kc * 2 + 1, lane, bh, bl);
        c1 = wmma_split(a1h[kc], a1l[kc], bh, bl, c1);
    }

    // ---- bias + relu -> LDS (C/D layout: lane L, vgpr r -> M=r+8*(L>>4), N=L&15)
    {
        const int col0 = lane & 15, col1 = col0 + 16;
        const float bi0 = ba[col0], bi1 = ba[col1];
#pragma unroll
        for (int r = 0; r < 8; ++r) {
            int m = r + 8 * hl;
            float y0 = c0[r] + bi0; ys[wave][m][col0] = y0 > 0.f ? y0 : 0.f;
            float y1 = c1[r] + bi1; ys[wave][m][col1] = y1 > 0.f ? y1 : 0.f;
        }
    }
    __syncthreads();

    // ---- layer 2 A fragment from ys (K = 32)
    v16bf a2h, a2l;
    {
        v8u hu, lu;
#pragma unroll
        for (int p = 0; p < 8; ++p) {
            int i = 2 * p;
            int k = 8 * hl + i + ((i & 8) ? 8 : 0);
            unsigned h, l;
            split_pack(ys[wave][mrow][k], ys[wave][mrow][k + 1], h, l);
            hu[p] = h; lu[p] = l;
        }
        a2h = __builtin_bit_cast(v16bf, hu);
        a2l = __builtin_bit_cast(v16bf, lu);
    }

    v8f d0 = {}, d1 = {};
    {
        v16bf bh, bl;
        load_bfrag_packed(fragB, 0, lane, bh, bl);
        d0 = wmma_split(a2h, a2l, bh, bl, d0);
        load_bfrag_packed(fragB, 1, lane, bh, bl);
        d1 = wmma_split(a2h, a2l, bh, bl, d1);
    }

    // ---- bias + relu + BN + outer relu, store
    {
        const int col0 = lane & 15, col1 = col0 + 16;
        const float s0 = gg[col0] * rsqrtf(var[col0] + 1e-3f);
        const float s1 = gg[col1] * rsqrtf(var[col1] + 1e-3f);
        const float t0 = beta[col0] - mu[col0] * s0;
        const float t1 = beta[col1] - mu[col1] * s1;
        const float bi0 = bbias[col0], bi1 = bbias[col1];
#pragma unroll
        for (int r = 0; r < 8; ++r) {
            int m = r + 8 * hl;
            int node = row0 + m;
            if (node < n) {
                float y0 = d0[r] + bi0; y0 = y0 > 0.f ? y0 : 0.f;
                y0 = y0 * s0 + t0;      y0 = y0 > 0.f ? y0 : 0.f;
                float y1 = d1[r] + bi1; y1 = y1 > 0.f ? y1 : 0.f;
                y1 = y1 * s1 + t1;      y1 = y1 > 0.f ? y1 : 0.f;
                Y[(size_t)node * 32 + col0] = y0;
                Y[(size_t)node * 32 + col1] = y1;
            }
        }
    }
}

// ---------------------------------------------------------------------------
// Per-graph sum pooling: pooled[batch[n]] += h[n]   (f32 atomics)
// ---------------------------------------------------------------------------
__global__ void pool_kernel(const float* __restrict__ H,
                            const int* __restrict__ batch,
                            float* __restrict__ pooled, int n) {
    long long tid = (long long)blockIdx.x * blockDim.x + threadIdx.x;
    if (tid >= (long long)n * 32) return;
    int node = (int)(tid >> 5);
    int d    = (int)(tid & 31);
    atomicAdd(&pooled[(size_t)batch[node] * 32 + d], H[tid]);
}

// ---------------------------------------------------------------------------
// Head: relu(pooled @ Wf1 + bf1) @ Wf2 + bf2 -> softmax.  256 rows: scalar.
// ---------------------------------------------------------------------------
__global__ void head_kernel(const float* __restrict__ pooled,
                            const float* __restrict__ Wf1, const float* __restrict__ bf1,
                            const float* __restrict__ Wf2, const float* __restrict__ bf2,
                            float* __restrict__ out, int g) {
    int i = blockIdx.x * blockDim.x + threadIdx.x;
    if (i >= g) return;
    float row[32];
#pragma unroll
    for (int k = 0; k < 32; ++k) row[k] = pooled[(size_t)i * 32 + k];
    float h[32];
#pragma unroll
    for (int c = 0; c < 32; ++c) {
        float acc = bf1[c];
#pragma unroll
        for (int k = 0; k < 32; ++k) acc += row[k] * Wf1[k * 32 + c];
        h[c] = acc > 0.f ? acc : 0.f;
    }
    float l0 = bf2[0], l1 = bf2[1];
#pragma unroll
    for (int k = 0; k < 32; ++k) { l0 += h[k] * Wf2[k * 2 + 0]; l1 += h[k] * Wf2[k * 2 + 1]; }
    float mx = l0 > l1 ? l0 : l1;
    float e0 = __expf(l0 - mx), e1 = __expf(l1 - mx);
    float inv = 1.f / (e0 + e1);
    out[(size_t)i * 2 + 0] = e0 * inv;
    out[(size_t)i * 2 + 1] = e1 * inv;
}

// ---------------------------------------------------------------------------
extern "C" void kernel_launch(void* const* d_in, const int* in_sizes, int n_in,
                              void* d_out, int out_size, void* d_ws, size_t ws_size,
                              hipStream_t stream) {
    const float* x     = (const float*)d_in[0];   // [N, 64]
    const int*   eidx  = (const int*)  d_in[1];   // [2, E]
    const int*   batch = (const int*)  d_in[2];   // [N]
    const float* W1a = (const float*)d_in[3];  const float* b1a = (const float*)d_in[4];
    const float* W1b = (const float*)d_in[5];  const float* b1b = (const float*)d_in[6];
    const float* g1  = (const float*)d_in[7];  const float* be1 = (const float*)d_in[8];
    const float* m1  = (const float*)d_in[9];  const float* v1  = (const float*)d_in[10];
    const float* W2a = (const float*)d_in[11]; const float* b2a = (const float*)d_in[12];
    const float* W2b = (const float*)d_in[13]; const float* b2b = (const float*)d_in[14];
    const float* g2  = (const float*)d_in[15]; const float* be2 = (const float*)d_in[16];
    const float* m2  = (const float*)d_in[17]; const float* v2  = (const float*)d_in[18];
    const float* Wf1 = (const float*)d_in[19]; const float* bf1 = (const float*)d_in[20];
    const float* Wf2 = (const float*)d_in[21]; const float* bf2 = (const float*)d_in[22];

    const int N = in_sizes[0] / 64;
    const int E = in_sizes[1] / 2;
    const int G = out_size / 2;
    const int* srcI = eidx;
    const int* dstI = eidx + E;

    // workspace layout (offsets 256B-aligned)
    char* ws = (char*)d_ws;
    float* agg1   = (float*)ws;                                   // N*64 f32
    float* h1     = (float*)(ws + (size_t)N * 64 * 4);            // N*32
    float* agg2   = (float*)(ws + (size_t)N * 96 * 4);            // N*32
    float* h2     = (float*)(ws + (size_t)N * 128 * 4);           // N*32
    float* pooled = (float*)(ws + (size_t)N * 160 * 4);           // G*32
    unsigned* wfrag = (unsigned*)(ws + (size_t)N * 160 * 4 + (size_t)G * 32 * 4);
    // packed weight fragments: each fragment = 512 dwords (2 KB)
    unsigned* f1a = wfrag;            // W1a: (64/32)*2 = 4 frags -> 2048 dwords
    unsigned* f1b = wfrag + 2048;     // W1b: 2 frags -> 1024
    unsigned* f2a = wfrag + 3072;     // W2a: 2 frags -> 1024
    unsigned* f2b = wfrag + 4096;     // W2b: 2 frags -> 1024
    (void)ws_size; (void)n_in;

    const int tiles  = (N + 15) / 16;
    const int mlpBlk = (tiles + MLP_WAVES - 1) / MLP_WAVES;

    // ---- pre-pack all weight fragments (tiny; once per launch) ----
    pack_wfrag_kernel<<<4, 32, 0, stream>>>(W1a, f1a);  // kc=0..1, nt=0..1
    pack_wfrag_kernel<<<2, 32, 0, stream>>>(W1b, f1b);  // kc=0,    nt=0..1
    pack_wfrag_kernel<<<2, 32, 0, stream>>>(W2a, f2a);
    pack_wfrag_kernel<<<2, 32, 0, stream>>>(W2b, f2b);

    // ---- conv1 ----
    {
        long long n4 = (long long)N * 16;                          // N*64/4
        copy4_kernel<<<(unsigned)((n4 + 255) / 256), 256, 0, stream>>>(
            (const float4*)x, (float4*)agg1, n4);
        long long tot = (long long)E * 4;
        edge_scatter_kernel<64><<<(unsigned)((tot + 255) / 256), 256, 0, stream>>>(
            x, srcI, dstI, agg1, E);
        gin_mlp_kernel<64><<<mlpBlk, MLP_WAVES * 32, 0, stream>>>(
            agg1, f1a, b1a, f1b, b1b, g1, be1, m1, v1, h1, N);
    }
    // ---- conv2 ----
    {
        long long n4 = (long long)N * 8;                           // N*32/4
        copy4_kernel<<<(unsigned)((n4 + 255) / 256), 256, 0, stream>>>(
            (const float4*)h1, (float4*)agg2, n4);
        long long tot = (long long)E * 2;
        edge_scatter_kernel<32><<<(unsigned)((tot + 255) / 256), 256, 0, stream>>>(
            h1, srcI, dstI, agg2, E);
        gin_mlp_kernel<32><<<mlpBlk, MLP_WAVES * 32, 0, stream>>>(
            agg2, f2a, b2a, f2b, b2b, g2, be2, m2, v2, h2, N);
    }
    // ---- pooling + head ----
    {
        int pn = G * 32;
        zero_kernel<<<(pn + 255) / 256, 256, 0, stream>>>(pooled, pn);
        long long tot = (long long)N * 32;
        pool_kernel<<<(unsigned)((tot + 255) / 256), 256, 0, stream>>>(h2, batch, pooled, N);
        head_kernel<<<(G + 255) / 256, 256, 0, stream>>>(
            pooled, Wf1, bf1, Wf2, bf2, (float*)d_out, G);
    }
}
